// LocalFeatureCorrelationLayer_64372969833079
// MI455X (gfx1250) — compile-verified, matrix operand order checked by reading, not run on hardware
//
#include <hip/hip_runtime.h>

// ---- problem constants (from reference setup_inputs) ----
#define B_DIM 16
#define C_DIM 256
#define H_DIM 128
#define W_DIM 128
#define HW    (H_DIM * W_DIM)
#define PATCH 9
#define PP    (PATCH * PATCH)   // 81

typedef __attribute__((ext_vector_type(16))) __bf16 v16bf;
typedef __attribute__((ext_vector_type(8)))  float  v8f;

union BFrag {
    v16bf v;
    unsigned int u[8];
};

// v_perm selector: pack {hi[31:16], lo[31:16]}  (bytes: lo.2, lo.3, hi.2, hi.3)
#define SEL_PACK 0x07060302u
// v_perm selector value 12 = constant 0x00 byte -> packed result is +0.0f pair
#define SEL_ZERO 0x0C0C0C0Cu

// round-to-nearest (half-up) f32->bf16 pair pack, with optional zeroing via sel
__device__ __forceinline__ unsigned int pack2_bf16(float lo, float hi, unsigned int sel) {
    unsigned int a = __float_as_uint(lo) + 0x8000u;
    unsigned int b = __float_as_uint(hi) + 0x8000u;
    return __builtin_amdgcn_perm(b, a, sel);
}

__global__ __launch_bounds__(32)
void lfc_wmma_kernel(const float* __restrict__ src,   // feature_source [B,C,H,W]
                     const float* __restrict__ tgt,   // feature_target [B,C,H,W]
                     float* __restrict__ out)         // [B,81,H,W]
{
    __shared__ float D_lds[2][16][16];     // two 16x16 f32 GEMM tiles
    __shared__ float corr_lds[16 * PP];    // relu'd corr, per pixel x 81
    __shared__ float inv_lds[16];

    const int lane = threadIdx.x;      // 0..31 (wave32)
    const int lp   = lane & 15;
    const int half = lane >> 4;

    // strip id -> (b, h, w-tile). 8 strips of 16 pixels per row.
    int sid = blockIdx.x;
    const int wt = sid & 7;  sid >>= 3;
    const int h  = sid & (H_DIM - 1); sid >>= 7;
    const int b  = sid;
    const int w0 = wt * 16;

    // uniform per-batch bases (keep all per-lane indexing as 32-bit ints so the
    // compiler emits saddr-form global_load_b32 with immediate channel offsets)
    const float* tb = tgt + (size_t)b * C_DIM * HW;
    const float* sb = src + (size_t)b * C_DIM * HW;

    // ---------------- load entire A (target strip) as bf16 fragments ----------------
    // A layout (16-bit, 16x32): lane m = lp; VGPR v holds K = (v&3)*2 + (v>>2)*16 + half*8, K+1
    unsigned int afrag[8][8];
    {
        const int ia = h * W_DIM + w0 + lp;   // per-lane 32-bit index
        #pragma unroll
        for (int ch = 0; ch < 8; ++ch) {
            const int c0 = ch * 32 + half * 8;
            #pragma unroll
            for (int v = 0; v < 8; ++v) {
                const int k = (v & 3) * 2 + (v >> 2) * 16;
                afrag[ch][v] = pack2_bf16(tb[ia + (c0 + k)     * HW],
                                          tb[ia + (c0 + k + 1) * HW],
                                          SEL_PACK);
            }
        }
    }

    // dh-invariant column geometry for the two B tiles:
    //   tile1: source cols w0-4+j   (j = lp)   -> only low clamp possible
    //   tile2: source cols w0+4+j   (j = lp)   -> only high clamp possible
    const int c1 = w0 - 4 + lp;
    const int c2 = w0 + 4 + lp;
    const unsigned int sel1 = (c1 >= 0)      ? SEL_PACK : SEL_ZERO;
    const unsigned int sel2 = (c2 < W_DIM)   ? SEL_PACK : SEL_ZERO;
    const int cc1 = (c1 < 0)        ? 0          : c1;   // clamped, always valid addr
    const int cc2 = (c2 >= W_DIM)   ? W_DIM - 1  : c2;

    float sumsq = 0.0f;   // meaningful in lanes 0..15 (one pixel each)

    // ---------------- displacement-row loop ----------------
    for (int dh = 0; dh < PATCH; ++dh) {
        const int srow = h + dh - PATCH / 2;
        const bool rowok = (srow >= 0) && (srow < H_DIM);

        if (rowok) {
            const int rbase = srow * W_DIM;
            const int i1 = rbase + cc1;
            const int i2 = rbase + cc2;

            v8f acc1 = {};
            v8f acc2 = {};

            #pragma unroll
            for (int ch = 0; ch < 8; ++ch) {
                const int c0 = ch * 32 + half * 8;
                BFrag a, b1, b2;
                #pragma unroll
                for (int v = 0; v < 8; ++v) {
                    const int k  = (v & 3) * 2 + (v >> 2) * 16;
                    const int o0 = (c0 + k)     * HW;
                    const int o1 = (c0 + k + 1) * HW;
                    b1.u[v] = pack2_bf16(sb[i1 + o0], sb[i1 + o1], sel1);
                    b2.u[v] = pack2_bf16(sb[i2 + o0], sb[i2 + o1], sel2);
                    a.u[v]  = afrag[ch][v];
                }
                // D = A x B + C  (bf16 in, f32 accumulate)
                acc1 = __builtin_amdgcn_wmma_f32_16x16x32_bf16(
                           false, a.v, false, b1.v, (short)0, acc1, false, false);
                acc2 = __builtin_amdgcn_wmma_f32_16x16x32_bf16(
                           false, a.v, false, b2.v, (short)0, acc2, false, false);
            }

            // C/D layout: VGPR r -> M = r + 8*half, N = lp
            #pragma unroll
            for (int r = 0; r < 8; ++r) {
                D_lds[0][r + 8 * half][lp] = acc1[r];
                D_lds[1][r + 8 * half][lp] = acc2[r];
            }
            __syncthreads();   // single-wave WG: s_nop + compiler memory fence

            // band extraction: out[i, dwi] with source col = w0 + i + dwi - 4
            if (lane < 16) {
                const int i = lane;
                #pragma unroll
                for (int dwi = 0; dwi < PATCH; ++dwi) {
                    const int j = i + dwi;
                    float val = (j <= 15) ? D_lds[0][i][j] : D_lds[1][i][j - 8];
                    val = fmaxf(val, 0.0f);              // ReLU
                    sumsq += val * val;
                    corr_lds[i * PP + dh * PATCH + dwi] = val;
                }
            }
            __syncthreads();
        } else {
            // source row out of image: zero padding -> corr = 0 for all 9 dw
            if (lane < 16) {
                #pragma unroll
                for (int dwi = 0; dwi < PATCH; ++dwi)
                    corr_lds[lane * PP + dh * PATCH + dwi] = 0.0f;
            }
            __syncthreads();
        }
    }

    // ---------------- normalize: x / max(||x||_2, 1e-12) ----------------
    if (lane < 16) {
        float n = sqrtf(sumsq);
        inv_lds[lane] = 1.0f / fmaxf(n, 1e-12f);
    }
    __syncthreads();

    const float inv = inv_lds[lp];
    float* ob = out + (size_t)b * PP * HW;          // uniform base
    const int obase = h * W_DIM + w0 + lp;          // per-lane 32-bit index
    // two displacement planes per iteration (lanes 0-15 -> d, lanes 16-31 -> d+1)
    for (int d0 = 0; d0 < PP - 1; d0 += 2) {
        const int d = d0 + half;
        ob[obase + d * HW] = corr_lds[lp * PP + d] * inv;
    }
    if (half == 0) {
        ob[obase + (PP - 1) * HW] = corr_lds[lp * PP + (PP - 1)] * inv;
    }
}

extern "C" void kernel_launch(void* const* d_in, const int* in_sizes, int n_in,
                              void* d_out, int out_size, void* d_ws, size_t ws_size,
                              hipStream_t stream) {
    const float* src = (const float*)d_in[0];   // feature_source
    const float* tgt = (const float*)d_in[1];   // feature_target
    float* out = (float*)d_out;

    const int strips = B_DIM * H_DIM * (W_DIM / 16);   // 16384
    lfc_wmma_kernel<<<strips, 32, 0, stream>>>(src, tgt, out);
}
